// WindowAttention_41197326303641
// MI455X (gfx1250) — matrix operand order repaired and checked by reading, not compiled, not run on hardware
//
#include <hip/hip_runtime.h>
#include <hip/hip_bf16.h>

typedef __attribute__((ext_vector_type(16))) __bf16        v16bf;
typedef __attribute__((ext_vector_type(8)))  float         v8f;
typedef __attribute__((ext_vector_type(8)))  unsigned int  v8u;

#define XS  264   // LDS row stride (shorts) for 64x256 bf16 tiles (conflict-free)
#define VTS 72    // LDS row stride for v-transposed [256][64]
#define PS  72    // LDS row stride for per-head probs [64][64]

static __device__ __forceinline__ unsigned short f2bf(float f) {
  unsigned int u = __builtin_bit_cast(unsigned int, f);
  u += 0x7FFFu + ((u >> 16) & 1u);          // round-to-nearest-even
  return (unsigned short)(u >> 16);
}

static __device__ __forceinline__ v16bf pack_frag(uint4 lo, uint4 hi) {
  v8u t;
  t[0] = lo.x; t[1] = lo.y; t[2] = lo.z; t[3] = lo.w;
  t[4] = hi.x; t[5] = hi.y; t[6] = hi.z; t[7] = hi.w;
  return __builtin_bit_cast(v16bf, t);
}

// A fragment 16x32 bf16: row = lane&15; lane-half picks K {0..7,16..23} vs {8..15,24..31}.
// base points at [row0][k0] of a row-major bf16 array with row stride `stride` shorts.
static __device__ __forceinline__ v16bf loadA(const unsigned short* base, int stride, int lane) {
  const unsigned short* p = base + (lane & 15) * stride + ((lane >> 4) << 3);
  uint4 lo = *(const uint4*)(p);
  uint4 hi = *(const uint4*)(p + 16);
  return pack_frag(lo, hi);
}

// B fragment 32x16 bf16 where B[k][n] = src[n*stride + k] (src row-major [N][K]).
// lane holds column lane&15; lane-half picks K {0..7,16..23} vs {8..15,24..31}.
static __device__ __forceinline__ v16bf loadB(const unsigned short* base, int stride, int lane) {
  const unsigned short* p = base + (lane & 15) * stride + ((lane >> 4) << 3);
  uint4 lo = *(const uint4*)(p);
  uint4 hi = *(const uint4*)(p + 16);
  return pack_frag(lo, hi);
}

// ---------------- DynamicPosBias MLP (225 rows, trivial) ----------------
static __device__ void ln_relu16(float* p, const float* g, const float* b) {
  float m = 0.f;
  for (int j = 0; j < 16; ++j) m += p[j];
  m *= (1.f / 16.f);
  float v = 0.f;
  for (int j = 0; j < 16; ++j) { float d = p[j] - m; v += d * d; }
  v *= (1.f / 16.f);
  float rs = rsqrtf(v + 1e-5f);
  for (int j = 0; j < 16; ++j) {
    float y = (p[j] - m) * rs * g[j] + b[j];
    p[j] = y > 0.f ? y : 0.f;
  }
}

__global__ void pos_mlp_kernel(const float* __restrict__ biases,
                               const float* __restrict__ pp_w, const float* __restrict__ pp_b,
                               const float* __restrict__ g1, const float* __restrict__ b1,
                               const float* __restrict__ w1, const float* __restrict__ bb1,
                               const float* __restrict__ g2, const float* __restrict__ b2,
                               const float* __restrict__ w2, const float* __restrict__ bb2,
                               const float* __restrict__ g3, const float* __restrict__ b3,
                               const float* __restrict__ w3, const float* __restrict__ bb3,
                               float* __restrict__ pos) {
  int r = blockIdx.x * blockDim.x + threadIdx.x;
  if (r >= 225) return;
  float p[16], t[16];
  float i0 = biases[2 * r], i1 = biases[2 * r + 1];
  for (int j = 0; j < 16; ++j) p[j] = pp_w[2 * j] * i0 + pp_w[2 * j + 1] * i1 + pp_b[j];
  ln_relu16(p, g1, b1);
  for (int j = 0; j < 16; ++j) {
    float s = bb1[j];
    for (int k = 0; k < 16; ++k) s += p[k] * w1[j * 16 + k];
    t[j] = s;
  }
  ln_relu16(t, g2, b2);
  for (int j = 0; j < 16; ++j) {
    float s = bb2[j];
    for (int k = 0; k < 16; ++k) s += t[k] * w2[j * 16 + k];
    p[j] = s;
  }
  ln_relu16(p, g3, b3);
  for (int h = 0; h < 8; ++h) {
    float s = bb3[h];
    for (int k = 0; k < 16; ++k) s += p[k] * w3[h * 16 + k];
    pos[r * 8 + h] = s;
  }
}

// bias[w][h][i][j] = pos[rel[i][j]][h] + mask[w][i][j]   (64*8*64*64 f32 = 8 MB, L2-resident)
__global__ void build_bias_kernel(const float* __restrict__ pos, const int* __restrict__ rel,
                                  const float* __restrict__ mask, float* __restrict__ bt) {
  int w = blockIdx.x >> 3;
  int h = blockIdx.x & 7;
  const float* mw = mask + (size_t)w * 4096;
  float* o = bt + (size_t)blockIdx.x * 4096;
  for (int e = threadIdx.x; e < 4096; e += blockDim.x)
    o[e] = pos[rel[e] * 8 + h] + mw[e];
}

// convert qkv_w [768x256] and proj_w [256x256] to bf16 once
__global__ void conv_w_kernel(const float* __restrict__ qw, const float* __restrict__ pw,
                              unsigned short* __restrict__ oq, unsigned short* __restrict__ op) {
  int i = blockIdx.x * blockDim.x + threadIdx.x;
  if (i < 768 * 256) oq[i] = f2bf(qw[i]);
  if (i < 256 * 256) op[i] = f2bf(pw[i]);
}

// ---------------- fused per-window attention ----------------
__global__ __launch_bounds__(256)
void fused_window_attn(const float* __restrict__ x,
                       const float* __restrict__ qkv_b,
                       const float* __restrict__ proj_b,
                       const unsigned short* __restrict__ qkvw,   // [768][256] bf16
                       const unsigned short* __restrict__ projw,  // [256][256] bf16
                       const float* __restrict__ biasT,           // [64][8][64][64]
                       float* __restrict__ out) {
  extern __shared__ unsigned short smem[];
  unsigned short* xb = smem;              // 64*XS
  unsigned short* qs = xb + 64 * XS;      // 64*XS  (q, pre-scaled)
  unsigned short* ks = qs + 64 * XS;      // 64*XS
  unsigned short* ob = ks + 64 * XS;      // 64*XS  (attention output, head-concat)
  unsigned short* vt = ob + 64 * XS;      // 256*VTS (v transposed: [c][token])
  unsigned short* pa = vt + 256 * VTS;    // 8*64*PS (per-head probs)

  const int tid = threadIdx.x;
  const int lane = tid & 31;
  const int wv = tid >> 5;                // wave id == head id
  const int b = blockIdx.x;
  const int hlf = lane >> 4;
  const int col = lane & 15;

  // ---- stage x[b] (64x256 f32) -> LDS bf16 ----
  const float4* xg = (const float4*)(x + (size_t)b * 16384);
  for (int e = tid; e < 4096; e += 256) {
    float4 v = xg[e];
    int row = e >> 6, c4 = (e & 63) << 2;
    unsigned short* d = xb + row * XS + c4;
    d[0] = f2bf(v.x); d[1] = f2bf(v.y); d[2] = f2bf(v.z); d[3] = f2bf(v.w);
  }
  __syncthreads();

  // ---- QKV GEMM: [64x256] @ W^T[256x768], 48 col-tiles striped over 8 waves ----
  const float qscale = 0.17677669529663687f;  // 32^-0.5
  for (int ct = wv; ct < 48; ct += 8) {
    const int n0 = ct << 4;
    if (ct + 8 < 48) __builtin_prefetch(qkvw + (size_t)(n0 + 128) * 256, 0, 3);
    v8f acc[4] = {};
    for (int kk = 0; kk < 8; ++kk) {
      const int k0 = kk << 5;
      v16bf bw = loadB(qkvw + (size_t)n0 * 256 + k0, 256, lane);
      for (int rt = 0; rt < 4; ++rt) {
        v16bf a = loadA(xb + (rt << 4) * XS + k0, XS, lane);
        acc[rt] = __builtin_amdgcn_wmma_f32_16x16x32_bf16(false, a, false, bw,
                                                          (short)0, acc[rt], false, false);
      }
    }
    const int c = n0 + col;
    const float bias = qkv_b[c];
    const int sec = c >> 8, cc = c & 255;
    for (int rt = 0; rt < 4; ++rt)
      for (int r = 0; r < 8; ++r) {
        int row = (rt << 4) + (hlf << 3) + r;
        float v = acc[rt][r] + bias;
        if (sec == 0)      qs[row * XS + cc] = f2bf(v * qscale);
        else if (sec == 1) ks[row * XS + cc] = f2bf(v);
        else               vt[cc * VTS + row] = f2bf(v);   // store v transposed
      }
  }
  __syncthreads();

  // ---- attention: wave wv owns head wv ----
  const int hc = wv << 5;
  unsigned short* pah = pa + wv * 64 * PS;
  const float* bt = biasT + (((size_t)(b & 63) << 3) + wv) * 4096;

  for (int rt = 0; rt < 4; ++rt) {
    v16bf aq = loadA(qs + (rt << 4) * XS + hc, XS, lane);  // q rows, K=d(0..31)
    v8f s[4];
    for (int cj = 0; cj < 4; ++cj) {
      v16bf bk = loadB(ks + (cj << 4) * XS + hc, XS, lane); // B[d][j]=k[j][d]
      v8f z = {};
      s[cj] = __builtin_amdgcn_wmma_f32_16x16x32_bf16(false, aq, false, bk,
                                                      (short)0, z, false, false);
    }
    // add (rpb + mask) bias
    for (int cj = 0; cj < 4; ++cj)
      for (int r = 0; r < 8; ++r) {
        int i = (rt << 4) + (hlf << 3) + r;
        s[cj][r] += bt[i * 64 + (cj << 4) + col];
      }
    // row softmax: rows live in (vgpr r, lane-half); reduce over cj then 16-lane half
    for (int r = 0; r < 8; ++r) {
      float mx = fmaxf(fmaxf(s[0][r], s[1][r]), fmaxf(s[2][r], s[3][r]));
      mx = fmaxf(mx, __shfl_xor(mx, 1, 32));
      mx = fmaxf(mx, __shfl_xor(mx, 2, 32));
      mx = fmaxf(mx, __shfl_xor(mx, 4, 32));
      mx = fmaxf(mx, __shfl_xor(mx, 8, 32));
      float sum = 0.f;
      for (int cj = 0; cj < 4; ++cj) {
        float e = __expf(s[cj][r] - mx);
        s[cj][r] = e;
        sum += e;
      }
      sum += __shfl_xor(sum, 1, 32);
      sum += __shfl_xor(sum, 2, 32);
      sum += __shfl_xor(sum, 4, 32);
      sum += __shfl_xor(sum, 8, 32);
      float inv = 1.0f / sum;
      int i = (rt << 4) + (hlf << 3) + r;
      for (int cj = 0; cj < 4; ++cj)
        pah[i * PS + (cj << 4) + col] = f2bf(s[cj][r] * inv);
    }
  }
  __syncthreads();

  // ---- P @ V : [64x64] @ [64x32] per head ----
  for (int rt = 0; rt < 4; ++rt) {
    v8f o2[2] = {};
    for (int kk = 0; kk < 2; ++kk) {
      v16bf ap = loadA(pah + (rt << 4) * PS + (kk << 5), PS, lane);
      for (int dt = 0; dt < 2; ++dt) {
        v16bf bv = loadB(vt + (hc + (dt << 4)) * VTS + (kk << 5), VTS, lane); // B[j][d]=v[j][d]
        o2[dt] = __builtin_amdgcn_wmma_f32_16x16x32_bf16(false, ap, false, bv,
                                                         (short)0, o2[dt], false, false);
      }
    }
    for (int dt = 0; dt < 2; ++dt)
      for (int r = 0; r < 8; ++r) {
        int row = (rt << 4) + (hlf << 3) + r;
        ob[row * XS + hc + (dt << 4) + col] = f2bf(o2[dt][r]);
      }
  }
  __syncthreads();

  // ---- output projection: [64x256] @ proj_w^T[256x256] -> global f32 ----
  float* og = out + (size_t)b * 16384;
  for (int ct = wv; ct < 16; ct += 8) {
    const int n0 = ct << 4;
    __builtin_prefetch(projw + (size_t)n0 * 256, 0, 3);
    v8f acc[4] = {};
    for (int kk = 0; kk < 8; ++kk) {
      const int k0 = kk << 5;
      v16bf bw = loadB(projw + (size_t)n0 * 256 + k0, 256, lane);
      for (int rt = 0; rt < 4; ++rt) {
        v16bf a = loadA(ob + (rt << 4) * XS + k0, XS, lane);
        acc[rt] = __builtin_amdgcn_wmma_f32_16x16x32_bf16(false, a, false, bw,
                                                          (short)0, acc[rt], false, false);
      }
    }
    const float bias = proj_b[n0 + col];
    for (int rt = 0; rt < 4; ++rt)
      for (int r = 0; r < 8; ++r) {
        int row = (rt << 4) + (hlf << 3) + r;
        og[row * 256 + n0 + col] = acc[rt][r] + bias;
      }
  }
}

extern "C" void kernel_launch(void* const* d_in, const int* in_sizes, int n_in,
                              void* d_out, int out_size, void* d_ws, size_t ws_size,
                              hipStream_t stream) {
  const float* x      = (const float*)d_in[0];
  // d_in[1] = k_v (unused by forward, faithful to reference)
  const float* mask   = (const float*)d_in[2];
  const float* biases = (const float*)d_in[3];
  const int*   rel    = (const int*)d_in[4];
  const float* qkv_w  = (const float*)d_in[5];
  const float* qkv_b  = (const float*)d_in[6];
  const float* proj_w = (const float*)d_in[7];
  const float* proj_b = (const float*)d_in[8];
  const float* pp_w   = (const float*)d_in[9];
  const float* pp_b   = (const float*)d_in[10];
  const float* ln1g   = (const float*)d_in[11];
  const float* ln1b   = (const float*)d_in[12];
  const float* l1w    = (const float*)d_in[13];
  const float* l1b    = (const float*)d_in[14];
  const float* ln2g   = (const float*)d_in[15];
  const float* ln2b   = (const float*)d_in[16];
  const float* l2w    = (const float*)d_in[17];
  const float* l2b    = (const float*)d_in[18];
  const float* ln3g   = (const float*)d_in[19];
  const float* ln3b   = (const float*)d_in[20];
  const float* l3w    = (const float*)d_in[21];
  const float* l3b    = (const float*)d_in[22];
  float* out = (float*)d_out;

  // workspace carve-up
  float* ws_pos  = (float*)d_ws;                               // 225*8 (pad to 4096 floats)
  float* ws_bias = ws_pos + 4096;                              // 64*8*64*64 floats (8 MB)
  unsigned short* ws_qkvw  = (unsigned short*)(ws_bias + (size_t)64 * 8 * 4096);
  unsigned short* ws_projw = ws_qkvw + 768 * 256;

  pos_mlp_kernel<<<1, 256, 0, stream>>>(biases, pp_w, pp_b,
                                        ln1g, ln1b, l1w, l1b,
                                        ln2g, ln2b, l2w, l2b,
                                        ln3g, ln3b, l3w, l3b, ws_pos);
  build_bias_kernel<<<512, 256, 0, stream>>>(ws_pos, rel, mask, ws_bias);
  conv_w_kernel<<<768, 256, 0, stream>>>(qkv_w, proj_w, ws_qkvw, ws_projw);

  const int lds_bytes = (4 * 64 * XS + 256 * VTS + 8 * 64 * PS) * (int)sizeof(unsigned short); // 245760
  hipFuncSetAttribute((const void*)fused_window_attn,
                      hipFuncAttributeMaxDynamicSharedMemorySize, lds_bytes);
  fused_window_attn<<<2048, 256, lds_bytes, stream>>>(x, qkv_b, proj_b,
                                                      ws_qkvw, ws_projw, ws_bias, out);
}
